// NNGAT_Net_17867063951385
// MI455X (gfx1250) — matrix operand order repaired and checked by reading, not compiled, not run on hardware
//
#include <hip/hip_runtime.h>
#include <hip/hip_bf16.h>
#include <math.h>

typedef __attribute__((ext_vector_type(16))) _Float16 v16h;
typedef __attribute__((ext_vector_type(8)))  _Float16 v8h;
typedef __attribute__((ext_vector_type(8)))  float    v8f;

// ---------------------------------------------------------------------------
// WMMA 16x16x32 f16 fragment layout (wave32), stored fragment-major in LDS so
// each lane reads its 16 halves as one contiguous 32-byte LDS load.
//   A (16x32 MxK): lane<16 row=lane holds K0..15; lane>=16 row=lane-16
//                  holds K16..31
//   B (32x16 KxN): lane<16 col=lane holds K0..15; lane>=16 col=lane-16 K16..31
// slot(A tile element r,c): g=c>>3 -> lane = r + ((g>>1)<<4), e = ((g&1)<<3)+(c&7)
// slot(B tile element k,c):           lane = c + ((k>>4)<<4), e = k&15
// ---------------------------------------------------------------------------
__device__ __forceinline__ int fragA_slot(int r, int c) {
  int g = c >> 3;
  return (r + ((g >> 1) << 4)) * 16 + ((g & 1) << 3) + (c & 7);
}

// Stage a 16x32 A tile (rows m0.., cols k0..) from row-major src[NR x K].
// 64 threads: thread t -> row r=t>>2, col-group g=t&3 (8 consecutive cols),
// which lands in 8 contiguous LDS halves -> one 16B LDS store.
template <int NR, int K>
__device__ __forceinline__ void stageA(const float* __restrict__ src, int m0,
                                       int k0, _Float16* sA, int t) {
  const int r = t >> 2, g = t & 3;
  const int gr = m0 + r;
  const int gc0 = k0 + g * 8;
  v8h tmp;
  if (gr < NR && gc0 + 8 <= K) {
    const float4* p4 = (const float4*)(src + (size_t)gr * K + gc0);
    float4 a = p4[0], b = p4[1];
    tmp[0] = (_Float16)a.x; tmp[1] = (_Float16)a.y;
    tmp[2] = (_Float16)a.z; tmp[3] = (_Float16)a.w;
    tmp[4] = (_Float16)b.x; tmp[5] = (_Float16)b.y;
    tmp[6] = (_Float16)b.z; tmp[7] = (_Float16)b.w;
  } else {
    const int cr = gr < NR ? gr : NR - 1;
#pragma unroll
    for (int u = 0; u < 8; ++u) {
      int gc = gc0 + u;
      int cc = gc < K ? gc : K - 1;
      float v = src[(size_t)cr * K + cc];           // always in-bounds
      tmp[u] = (_Float16)((gr < NR && gc < K) ? v : 0.f);
    }
  }
  const int slot = (r + ((g >> 1) << 4)) * 16 + ((g & 1) << 3);
  *(v8h*)(sA + slot) = tmp;
}

// Stage a 32x32 B tile (rows k0.., all 32 cols = 2 wave tiles) from row-major
// src[K x 32] into fragment-major sB[2*512]. 64 threads: thread t -> column
// c_full=t>>1, K-half=t&1 (16 rows) -> one contiguous 32B LDS store.
template <int K>
__device__ __forceinline__ void stageB32(const float* __restrict__ src, int k0,
                                         _Float16* sB, int t) {
  const int c_full = t >> 1;
  const int krb = (t & 1) * 16;
  v16h tmp;
#pragma unroll
  for (int u = 0; u < 16; ++u) {
    int gk = k0 + krb + u;
    int ck = gk < K ? gk : K - 1;
    float v = src[(size_t)ck * 32 + c_full];        // always in-bounds
    tmp[u] = (_Float16)(gk < K ? v : 0.f);
  }
  *(v16h*)(sB + (c_full >> 4) * 512 + ((c_full & 15) + krb) * 16) = tmp;
}

// ---------------- GEMM: C[b] (NR x 32) = A[b] (NR x K) @ W (K x 32) ----------
// grid(ceil(NR/16), B), block 64 (2 waves; wave w owns output cols w*16..+15)
template <int NR, int K>
__global__ void gemm_t(const float* __restrict__ A, const float* __restrict__ W,
                       float* __restrict__ C) {
  const int b = blockIdx.y;
  const int m0 = blockIdx.x * 16;
  const int t = threadIdx.x, lane = t & 31, wave = t >> 5;
  __shared__ __align__(32) _Float16 sA[512];
  __shared__ __align__(32) _Float16 sB[1024];
  const float* Ab = A + (size_t)b * NR * K;
  v8f acc = {};
#pragma unroll
  for (int k0 = 0; k0 < K; k0 += 32) {
    stageA<NR, K>(Ab, m0, k0, sA, t);
    stageB32<K>(W, k0, sB, t);
    __syncthreads();
    v16h fa = *(const v16h*)(sA + lane * 16);
    v16h fb = *(const v16h*)(sB + wave * 512 + lane * 16);
    acc = __builtin_amdgcn_wmma_f32_16x16x32_f16(false, fa, false, fb, (short)0,
                                                 acc, false, false);
    __syncthreads();
  }
  const int col = wave * 16 + (lane & 15);
  const int rbase = (lane >> 4) << 3;
#pragma unroll
  for (int e = 0; e < 8; ++e) {
    int row = m0 + rbase + e;
    if (row < NR) C[(size_t)b * NR * 32 + (size_t)row * 32 + col] = acc[e];
  }
}

// ------------- es = h . a_s, ed = h . a_d (per node) -------------------------
__global__ void esed_k(const float* __restrict__ h, const float* __restrict__ as,
                       const float* __restrict__ ad, float* __restrict__ es,
                       float* __restrict__ ed, int total) {
  int i = blockIdx.x * blockDim.x + threadIdx.x;
  if (i >= total) return;
  const float4* hr = (const float4*)(h + (size_t)i * 32);
  float s = 0.f, d = 0.f;
#pragma unroll
  for (int c = 0; c < 8; ++c) {
    float4 v = hr[c];
    s += v.x * as[c * 4] + v.y * as[c * 4 + 1] + v.z * as[c * 4 + 2] + v.w * as[c * 4 + 3];
    d += v.x * ad[c * 4] + v.y * ad[c * 4 + 1] + v.z * ad[c * 4 + 2] + v.w * ad[c * 4 + 3];
  }
  es[i] = s;
  ed[i] = d;
}

// ------------- GAT attention: g = softmax(mask(leaky(ed_i+es_j))) @ h + bias -
// grid(ceil(N/16), B), block 64. mask true iff mw[b,i,j] != 0 or i==j.
template <int N>
__global__ void attn_t(const float* __restrict__ h, const float* __restrict__ es,
                       const float* __restrict__ ed, const float* __restrict__ mw,
                       const float* __restrict__ bias, float* __restrict__ g) {
  constexpr int NT = (N + 31) / 32;          // k-tiles
  const int b = blockIdx.y;
  const int i0 = blockIdx.x * 16;
  const int t = threadIdx.x, lane = t & 31, wave = t >> 5;
  __shared__ __align__(32) _Float16 sAl[NT * 512];  // alpha, fragment-major
  __shared__ __align__(32) _Float16 sH[1024];       // h rows tile
  __shared__ float sRed[64];
  __shared__ float sMax[16], sSum[16];

  for (int idx = t; idx < NT * 512; idx += 64) sAl[idx] = (_Float16)0.f;

  const int r = t >> 2, p = t & 3;
  const int i = i0 + r;
  const bool vrow = (i < N);
  const float edi = vrow ? ed[(size_t)b * N + i] : 0.f;
  const float* mrow = mw + (size_t)b * N * N + (size_t)(vrow ? i : 0) * N;
  const float* esb = es + (size_t)b * N;

  // pass 1: row max of masked leaky-relu logits
  float lm = -1e30f;
  if (vrow) {
    for (int j = p; j < N; j += 4) {
      __builtin_prefetch(mrow + j + 64, 0, 1);
      if (mrow[j] != 0.f || j == i) {
        float l = edi + esb[j];
        l = l > 0.f ? l : 0.2f * l;
        lm = fmaxf(lm, l);
      }
    }
  }
  sRed[t] = lm;
  __syncthreads();
  if (p == 0) {
    float m0 = fmaxf(fmaxf(sRed[t], sRed[t + 1]), fmaxf(sRed[t + 2], sRed[t + 3]));
    sMax[r] = vrow ? m0 : 0.f;
  }
  __syncthreads();
  const float mi = sMax[r];

  // pass 2: exp + row sum, store unnormalized alpha into fragment slots
  float ls = 0.f;
  if (vrow) {
    for (int j = p; j < N; j += 4) {
      float e = 0.f;
      if (mrow[j] != 0.f || j == i) {
        float l = edi + esb[j];
        l = l > 0.f ? l : 0.2f * l;
        e = __expf(l - mi);
      }
      sAl[(j >> 5) * 512 + fragA_slot(r, j & 31)] = (_Float16)e;
      ls += e;
    }
  }
  sRed[t] = ls;
  __syncthreads();
  if (p == 0) sSum[r] = sRed[t] + sRed[t + 1] + sRed[t + 2] + sRed[t + 3];
  __syncthreads();
  if (vrow) {
    float inv = 1.f / sSum[r];
    for (int j = p; j < N; j += 4) {
      int s = (j >> 5) * 512 + fragA_slot(r, j & 31);
      sAl[s] = (_Float16)((float)sAl[s] * inv);
    }
  }
  __syncthreads();

  // alpha[16 x N] @ h[N x 32] via WMMA
  v8f acc = {};
#pragma unroll
  for (int kt = 0; kt < NT; ++kt) {
    stageB32<N>(h + (size_t)b * N * 32, kt * 32, sH, t);
    __syncthreads();
    v16h fa = *(const v16h*)(sAl + kt * 512 + lane * 16);
    v16h fb = *(const v16h*)(sH + wave * 512 + lane * 16);
    acc = __builtin_amdgcn_wmma_f32_16x16x32_f16(false, fa, false, fb, (short)0,
                                                 acc, false, false);
    __syncthreads();
  }
  const int col = wave * 16 + (lane & 15);
  const int rbase = (lane >> 4) << 3;
  const float bc = bias[col];
#pragma unroll
  for (int e = 0; e < 8; ++e) {
    int row = i0 + rbase + e;
    if (row < N) g[(size_t)b * N * 32 + (size_t)row * 32 + col] = acc[e] + bc;
  }
}

// ------------- TopK pooling scores + rank-select (jax.lax.top_k semantics) ---
__global__ void topk_k(const float* __restrict__ g, const float* __restrict__ pw,
                       int* __restrict__ idx, float* __restrict__ vals, int n, int k) {
  const int b = blockIdx.x;
  const int t = threadIdx.x;          // 256 threads
  __shared__ float sc[224];
  float nrm = 0.f;
#pragma unroll
  for (int c = 0; c < 32; ++c) nrm += pw[c] * pw[c];
  const float inv = 1.f / (sqrtf(nrm) + 1e-16f);
  if (t < n) {
    const float* gr = g + (size_t)b * n * 32 + (size_t)t * 32;
    float dot = 0.f;
#pragma unroll
    for (int c = 0; c < 32; ++c) dot += gr[c] * pw[c];
    float x = dot * inv;
    sc[t] = 1.f / (1.f + __expf(-x));
  }
  __syncthreads();
  if (t < n) {
    float si = sc[t];
    int rank = 0;
    for (int j = 0; j < n; ++j) {
      float sj = sc[j];
      rank += (sj > si) || (sj == si && j < t);
    }
    if (rank < k) {
      idx[(size_t)b * k + rank] = t;
      vals[(size_t)b * k + rank] = si;
    }
  }
}

// ------------- gather pooled features (scaled by vals) and sub-adjacency -----
__global__ void gather_k(const float* __restrict__ g, const float* __restrict__ adjw,
                         const int* __restrict__ idx, const float* __restrict__ vals,
                         float* __restrict__ hp, float* __restrict__ aw, int n, int k) {
  const int b = blockIdx.x;
  const int t = threadIdx.x;          // 256 threads
  const int* ib = idx + (size_t)b * k;
  const float* vb = vals + (size_t)b * k;
  for (int e = t; e < k * 32; e += 256) {
    int r = e >> 5, c = e & 31;
    hp[(size_t)b * k * 32 + e] = g[(size_t)b * n * 32 + (size_t)ib[r] * 32 + c] * vb[r];
  }
  if (aw) {
    for (int e = t; e < k * k; e += 256) {
      int r = e / k, c = e % k;
      aw[(size_t)b * k * k + e] = adjw[(size_t)b * n * n + (size_t)ib[r] * n + ib[c]];
    }
  }
}

// ------------- readout: z[b, off..off+63] = [max_r hp, mean_r hp] ------------
__global__ void pool_k(const float* __restrict__ hp, float* __restrict__ z,
                       int k, int off) {
  const int b = blockIdx.x;
  const int d = threadIdx.x;          // 32 threads
  const float* base = hp + (size_t)b * k * 32 + d;
  float m = -1e30f, s = 0.f;
  for (int r = 0; r < k; ++r) { float v = base[r * 32]; m = fmaxf(m, v); s += v; }
  z[(size_t)b * 128 + off + d] = m;
  z[(size_t)b * 128 + off + 32 + d] = s / (float)k;
}

// ------------- a2 = (aw+I)(aw+I) * (1-I), 100x100x100 via WMMA ---------------
// grid(7, B), block 224 (7 waves; wave w owns output cols w*16..w*16+15)
__global__ void a2_k(const float* __restrict__ aw, float* __restrict__ a2) {
  const int b = blockIdx.y;
  const int m0 = blockIdx.x * 16;
  const int t = threadIdx.x, lane = t & 31, wave = t >> 5;
  __shared__ __align__(32) _Float16 sA[512];
  __shared__ __align__(32) _Float16 sB[7 * 512];
  const float* awb = aw + (size_t)b * 100 * 100;
  v8f acc = {};
#pragma unroll
  for (int k0 = 0; k0 < 128; k0 += 32) {
    if (t < 64) {                         // A tile: rows m0.. of (aw+I), K k0..
      const int r = t >> 2, g = t & 3;
      const int gm = m0 + r;
      const int gc0 = k0 + g * 8;
      const int cm = gm < 100 ? gm : 99;
      v8h tmp;
#pragma unroll
      for (int u = 0; u < 8; ++u) {
        int gc = gc0 + u;
        int cc = gc < 100 ? gc : 99;
        float v = awb[cm * 100 + cc];     // always in-bounds
        v = (gm < 100 && gc < 100) ? v : 0.f;
        if (gm == gc) v += 1.f;
        tmp[u] = (_Float16)v;
      }
      *(v8h*)(sA + (r + ((g >> 1) << 4)) * 16 + ((g & 1) << 3)) = tmp;
    }
    {                                     // B tile: rows k0..k0+31, cols 0..111
      const int c_full = t >> 1;          // 0..111
      const int krb = (t & 1) * 16;
      v16h tmp;
#pragma unroll
      for (int u = 0; u < 16; ++u) {
        int gk = k0 + krb + u;
        int ck = gk < 100 ? gk : 99;
        int cc = c_full < 100 ? c_full : 99;
        float v = awb[ck * 100 + cc];     // always in-bounds
        v = (gk < 100 && c_full < 100) ? v : 0.f;
        if (gk == c_full) v += 1.f;
        tmp[u] = (_Float16)v;
      }
      *(v16h*)(sB + (c_full >> 4) * 512 + ((c_full & 15) + krb) * 16) = tmp;
    }
    __syncthreads();
    v16h fa = *(const v16h*)(sA + lane * 16);
    v16h fb = *(const v16h*)(sB + wave * 512 + lane * 16);
    acc = __builtin_amdgcn_wmma_f32_16x16x32_f16(false, fa, false, fb, (short)0,
                                                 acc, false, false);
    __syncthreads();
  }
  const int col = wave * 16 + (lane & 15);
  const int rbase = (lane >> 4) << 3;
#pragma unroll
  for (int e = 0; e < 8; ++e) {
    int row = m0 + rbase + e;
    if (row < 100 && col < 100)
      a2[(size_t)b * 10000 + row * 100 + col] = (row == col) ? 0.f : acc[e];
  }
}

// ------------- MLP head: fc1->relu->bn, fc2->relu->bn, fc3->log_softmax ------
__global__ void mlp_k(const float* __restrict__ z, const float* __restrict__ w1,
                      const float* __restrict__ b1, const float* __restrict__ w2,
                      const float* __restrict__ b2, const float* __restrict__ w3,
                      const float* __restrict__ b3, const float* __restrict__ g4,
                      const float* __restrict__ bb4, const float* __restrict__ g5,
                      const float* __restrict__ bb5, float* __restrict__ out) {
  const int b = blockIdx.x;
  const int t = threadIdx.x;          // 32 threads
  __shared__ float zz[128], z1[32], z2[8], o[2];
  for (int e = t; e < 128; e += 32) zz[e] = z[(size_t)b * 128 + e];
  __syncthreads();
  const float ibn = rsqrtf(1.f + 1e-5f);
  {
    float a = b1[t];
    for (int f = 0; f < 128; ++f) a += zz[f] * w1[f * 32 + t];
    a = fmaxf(a, 0.f);
    z1[t] = g4[t] * a * ibn + bb4[t];
  }
  __syncthreads();
  if (t < 8) {
    float a = b2[t];
#pragma unroll
    for (int f = 0; f < 32; ++f) a += z1[f] * w2[f * 8 + t];
    a = fmaxf(a, 0.f);
    z2[t] = g5[t] * a * ibn + bb5[t];
  }
  __syncthreads();
  if (t < 2) {
    float a = b3[t];
#pragma unroll
    for (int f = 0; f < 8; ++f) a += z2[f] * w3[f * 2 + t];
    o[t] = a;
  }
  __syncthreads();
  if (t < 2) {
    float m = fmaxf(o[0], o[1]);
    float lse = m + logf(__expf(o[0] - m) + __expf(o[1] - m));
    out[(size_t)b * 2 + t] = o[t] - lse;
  }
}

extern "C" void kernel_launch(void* const* d_in, const int* in_sizes, int n_in,
                              void* d_out, int out_size, void* d_ws, size_t ws_size,
                              hipStream_t stream) {
  (void)in_sizes; (void)n_in; (void)out_size; (void)ws_size;
  const float* x    = (const float*)d_in[0];
  const float* adj  = (const float*)d_in[1];
  const float* W1   = (const float*)d_in[2];
  const float* a1s  = (const float*)d_in[3];
  const float* a1d  = (const float*)d_in[4];
  const float* b1   = (const float*)d_in[5];
  const float* W2   = (const float*)d_in[6];
  const float* a2s  = (const float*)d_in[7];
  const float* a2d  = (const float*)d_in[8];
  const float* b2   = (const float*)d_in[9];
  const float* pw1  = (const float*)d_in[10];
  const float* pw2  = (const float*)d_in[11];
  const float* fc1w = (const float*)d_in[12];
  const float* fc1b = (const float*)d_in[13];
  const float* fc2w = (const float*)d_in[14];
  const float* fc2b = (const float*)d_in[15];
  const float* fc3w = (const float*)d_in[16];
  const float* fc3b = (const float*)d_in[17];
  const float* bn4g = (const float*)d_in[18];
  const float* bn4b = (const float*)d_in[19];
  const float* bn5g = (const float*)d_in[20];
  const float* bn5b = (const float*)d_in[21];

  float* out = (float*)d_out;
  float* s1  = out + 1024;                      // [512,100]
  float* s2  = out + 52224;                     // [512,50]

  const int B = 512;
  float* ws = (float*)d_ws;
  size_t o = 0;
  float* hbuf = ws + o; o += (size_t)B * 200 * 32;   // h1 / h2
  float* gbuf = ws + o; o += (size_t)B * 200 * 32;   // g1 / g2
  float* es   = ws + o; o += (size_t)B * 200;
  float* ed   = ws + o; o += (size_t)B * 200;
  float* hp   = ws + o; o += (size_t)B * 100 * 32;   // hp1 / hp2
  float* aw   = ws + o; o += (size_t)B * 100 * 100;
  float* a2   = ws + o; o += (size_t)B * 100 * 100;
  float* zb   = ws + o; o += (size_t)B * 128;
  int* idx1 = (int*)(ws + o); o += (size_t)B * 100;
  int* idx2 = (int*)(ws + o); o += (size_t)B * 50;

  // ----- stage 1: GAT1 on (x, adj) -----
  gemm_t<200, 200><<<dim3(13, B), 64, 0, stream>>>(x, W1, hbuf);
  esed_k<<<(B * 200 + 255) / 256, 256, 0, stream>>>(hbuf, a1s, a1d, es, ed, B * 200);
  attn_t<200><<<dim3(13, B), 64, 0, stream>>>(hbuf, es, ed, adj, b1, gbuf);
  topk_k<<<B, 256, 0, stream>>>(gbuf, pw1, idx1, s1, 200, 100);
  gather_k<<<B, 256, 0, stream>>>(gbuf, adj, idx1, s1, hp, aw, 200, 100);
  pool_k<<<B, 32, 0, stream>>>(hp, zb, 100, 0);
  a2_k<<<dim3(7, B), 224, 0, stream>>>(aw, a2);

  // ----- stage 2: GAT2 on (hp, a2) -----
  gemm_t<100, 32><<<dim3(7, B), 64, 0, stream>>>(hp, W2, hbuf);
  esed_k<<<(B * 100 + 255) / 256, 256, 0, stream>>>(hbuf, a2s, a2d, es, ed, B * 100);
  attn_t<100><<<dim3(7, B), 64, 0, stream>>>(hbuf, es, ed, a2, b2, gbuf);
  topk_k<<<B, 256, 0, stream>>>(gbuf, pw2, idx2, s2, 100, 50);
  gather_k<<<B, 256, 0, stream>>>(gbuf, nullptr, idx2, s2, hp, nullptr, 100, 50);
  pool_k<<<B, 32, 0, stream>>>(hp, zb, 50, 64);

  // ----- head -----
  mlp_k<<<B, 32, 0, stream>>>(zb, fc1w, fc1b, fc2w, fc2b, fc3w, fc3b,
                              bn4g, bn4b, bn5g, bn5b, out);
}
